// CrossAttentionGating_23398981829196
// MI455X (gfx1250) — matrix operand order repaired and checked by reading, not compiled, not run on hardware
//
#include <hip/hip_runtime.h>
#include <hip/hip_fp16.h>
#include <math.h>

// Problem constants (from reference): B=8, TQ=512, TK=64, D=512
#define DD   512
#define BB   8
#define TQ_  512
#define TK_  64

typedef __attribute__((ext_vector_type(16))) _Float16 v16h;
typedef __attribute__((ext_vector_type(8)))  _Float16 v8h;
typedef __attribute__((ext_vector_type(4)))  _Float16 v4h;
typedef __attribute__((ext_vector_type(8)))  float    v8f;

#if __has_builtin(__builtin_amdgcn_tanhf)
__device__ __forceinline__ float fast_tanh(float x) { return __builtin_amdgcn_tanhf(x); }
#else
__device__ __forceinline__ float fast_tanh(float x) { return tanhf(x); }
#endif

#if __has_builtin(__builtin_amdgcn_global_load_async_to_lds_b128) && \
    __has_builtin(__builtin_amdgcn_s_wait_asynccnt)
#define ASYNC_STAGE 1
typedef int v4i_ __attribute__((vector_size(16)));
typedef __attribute__((address_space(1))) v4i_ as1_v4i;
typedef __attribute__((address_space(3))) v4i_ as3_v4i;
// Copy 16B global -> LDS via ASYNCcnt-tracked async load.
__device__ __forceinline__ void async_cp16(const _Float16* g, _Float16* l) {
  _Float16* gm = const_cast<_Float16*>(g);
  __builtin_amdgcn_global_load_async_to_lds_b128((as1_v4i*)gm, (as3_v4i*)l,
                                                 0, 0);
}
#else
#define ASYNC_STAGE 0
#endif

__device__ __forceinline__ float wave_sum32(float x) {
#pragma unroll
  for (int off = 16; off > 0; off >>= 1) x += __shfl_xor(x, off, 32);
  return x;
}
__device__ __forceinline__ float wave_max32(float x) {
#pragma unroll
  for (int off = 16; off > 0; off >>= 1) x = fmaxf(x, __shfl_xor(x, off, 32));
  return x;
}

// ---------------------------------------------------------------------------
// fp32 -> f16 bulk conversion (one-time pre-pass; float4 -> 4 halves)
// ---------------------------------------------------------------------------
__global__ __launch_bounds__(256) void cvt_f32_to_f16(
    const float* __restrict__ in, _Float16* __restrict__ out, int n4) {
  int i = blockIdx.x * 256 + threadIdx.x;
  if (i < n4) {
    float4 f = reinterpret_cast<const float4*>(in)[i];
    v4h h;
    h[0] = (_Float16)f.x; h[1] = (_Float16)f.y;
    h[2] = (_Float16)f.z; h[3] = (_Float16)f.w;
    reinterpret_cast<v4h*>(out)[i] = h;
  }
}

// ---------------------------------------------------------------------------
// OUT[m,n] = [gate[m,n] *] act( sum_k X[m,k]*W[n*wstride+coff+k] [+ bias[n]] )
// X, W are pre-converted f16.  Block = 256 threads (8 waves), tile = 64x64.
// Wave w -> (wm=w>>1) 16 rows, (wn=w&1) 32 cols = two 16x16x32 WMMA frags.
// Staging: one b128 per thread per tile, via async global->LDS if available.
// ---------------------------------------------------------------------------
template <int HAS_BIAS, int ACT_SIG, int HAS_GATE>
__global__ __launch_bounds__(256, 2) void gemm_wmma(
    const _Float16* __restrict__ X, int lda, const _Float16* __restrict__ W,
    int wstride, int coff, const float* __restrict__ bias,
    const float* __restrict__ gate, float* __restrict__ out) {
  __shared__ _Float16 As[64][40];  // 64 rows x 32 K, 80B row stride (16B mult)
  __shared__ _Float16 Bs[64][40];

  const int tid  = threadIdx.x;
  const int wave = tid >> 5, lane = tid & 31;
  const int wm = wave >> 1, wn = wave & 1;
  const int hl = lane >> 4, mr = lane & 15;
  const int m0 = blockIdx.x * 64, n0 = blockIdx.y * 64;

  const int r  = tid >> 2;        // staging row 0..63
  const int c8 = (tid & 3) * 8;   // staging column (halves), 16B granules

  v8f acc[2] = {};

  for (int k0 = 0; k0 < DD; k0 += 32) {
    const _Float16* gA = X + (size_t)(m0 + r) * lda + k0 + c8;
    const _Float16* gB = W + (size_t)(n0 + r) * wstride + coff + k0 + c8;
#if ASYNC_STAGE
    async_cp16(gA, &As[r][c8]);
    async_cp16(gB, &Bs[r][c8]);
    __builtin_amdgcn_s_wait_asynccnt(0);
#else
    *reinterpret_cast<v8h*>(&As[r][c8]) = *reinterpret_cast<const v8h*>(gA);
    *reinterpret_cast<v8h*>(&Bs[r][c8]) = *reinterpret_cast<const v8h*>(gB);
#endif
    if (k0 + 32 < DD) {  // pull next slab toward L2 (global_prefetch_b8)
      __builtin_prefetch(gA + 32, 0, 1);
      __builtin_prefetch(gB + 32, 0, 1);
    }
    __syncthreads();

    // Fragments use contiguous K order, same for A and B:
    // A 16x32: lane<16 -> M=lane, K=j ; lane>=16 -> M=lane-16, K=16+j
    // B 32x16: lane<16 -> N=lane, K=j ; lane>=16 -> N=lane-16, K=16+j
    v16h a, b0, b1;
#pragma unroll
    for (int j = 0; j < 16; ++j) {
      int ka = 16 * hl + j;
      a[j] = As[wm * 16 + mr][ka];
      int kb = 16 * hl + j;
      b0[j] = Bs[wn * 32 + mr][kb];
      b1[j] = Bs[wn * 32 + 16 + mr][kb];
    }
    acc[0] = __builtin_amdgcn_wmma_f32_16x16x32_f16(false, a, false, b0,
                                                    (short)0, acc[0], false, false);
    acc[1] = __builtin_amdgcn_wmma_f32_16x16x32_f16(false, a, false, b1,
                                                    (short)0, acc[1], false, false);
    __syncthreads();
  }

  // C/D layout: elem r -> M = 8*hl + r, N = mr
#pragma unroll
  for (int s = 0; s < 2; ++s) {
    int n = n0 + wn * 32 + s * 16 + mr;
    float bv = HAS_BIAS ? bias[n] : 0.f;
#pragma unroll
    for (int rr = 0; rr < 8; ++rr) {
      int m = m0 + wm * 16 + 8 * hl + rr;
      float val = acc[s][rr] + bv;
      if (ACT_SIG) val = 1.f / (1.f + __expf(-val));
      if (HAS_GATE) val *= gate[m * DD + n];
      out[m * DD + n] = val;
    }
  }
}

// ---------------------------------------------------------------------------
// Fused score / mask / softmax / context.  One wave32 per (b, q) row.
// Per-lane d = lane*4 + 128*i (i<4): all VMEM in the hot loop is b128.
// Also emits ctx as f16 (A-operand of the final gated GEMM).
// ---------------------------------------------------------------------------
__global__ __launch_bounds__(256) void score_softmax_ctx(
    const float* __restrict__ qp, const float* __restrict__ kb,
    const float* __restrict__ text, const float* __restrict__ vvec,
    const int* __restrict__ alen, const int* __restrict__ tlen,
    float* __restrict__ ctx, _Float16* __restrict__ ctx_h) {
  __shared__ float attn_s[8][64];
  const int wave = threadIdx.x >> 5, lane = threadIdx.x & 31;
  const int row = blockIdx.x * 8 + wave;  // 0 .. B*TQ-1
  const int b = row >> 9, q = row & (TQ_ - 1);
  const int dbase = lane * 4;

  const float* qrow = qp + (size_t)row * DD;
  float4 qv[4], vv[4];
#pragma unroll
  for (int i = 0; i < 4; ++i) {
    qv[i] = *reinterpret_cast<const float4*>(qrow + dbase + 128 * i);
    vv[i] = *reinterpret_cast<const float4*>(vvec + dbase + 128 * i);
  }

  const bool qok = q < alen[b];
  const int tl = tlen[b];
  const float* kbb = kb + (size_t)b * TK_ * DD;

  float sc0 = 0.f, sc1 = 0.f;  // scores for k = lane and k = lane + 32
  for (int k = 0; k < TK_; ++k) {
    const float* kr = kbb + (size_t)k * DD;
    float acc = 0.f;
#pragma unroll
    for (int i = 0; i < 4; ++i) {
      float4 kv = *reinterpret_cast<const float4*>(kr + dbase + 128 * i);
      acc += fast_tanh(qv[i].x + kv.x) * vv[i].x;
      acc += fast_tanh(qv[i].y + kv.y) * vv[i].y;
      acc += fast_tanh(qv[i].z + kv.z) * vv[i].z;
      acc += fast_tanh(qv[i].w + kv.w) * vv[i].w;
    }
    acc = wave_sum32(acc);
    float s = (qok && (k < tl)) ? acc : -1e10f;
    if ((k & 31) == lane) {
      if (k < 32) sc0 = s; else sc1 = s;
    }
  }

  float mx  = wave_max32(fmaxf(sc0, sc1));
  float p0  = __expf(sc0 - mx), p1 = __expf(sc1 - mx);
  float inv = 1.f / wave_sum32(p0 + p1);
  attn_s[wave][lane]      = p0 * inv;
  attn_s[wave][lane + 32] = p1 * inv;
  __syncthreads();

  const float* trow = text + (size_t)b * TK_ * DD;
  float4 c[4];
#pragma unroll
  for (int i = 0; i < 4; ++i) c[i] = make_float4(0.f, 0.f, 0.f, 0.f);
  for (int k = 0; k < TK_; ++k) {
    float ak = attn_s[wave][k];
    const float* tk = trow + (size_t)k * DD;
#pragma unroll
    for (int i = 0; i < 4; ++i) {
      float4 tv = *reinterpret_cast<const float4*>(tk + dbase + 128 * i);
      c[i].x += ak * tv.x; c[i].y += ak * tv.y;
      c[i].z += ak * tv.z; c[i].w += ak * tv.w;
    }
  }
  float* crow = ctx + (size_t)row * DD;
  _Float16* hrow = ctx_h + (size_t)row * DD;
#pragma unroll
  for (int i = 0; i < 4; ++i) {
    *reinterpret_cast<float4*>(crow + dbase + 128 * i) = c[i];
    v4h h;
    h[0] = (_Float16)c[i].x; h[1] = (_Float16)c[i].y;
    h[2] = (_Float16)c[i].z; h[3] = (_Float16)c[i].w;
    *reinterpret_cast<v4h*>(hrow + dbase + 128 * i) = h;
  }
}

extern "C" void kernel_launch(void* const* d_in, const int* in_sizes, int n_in,
                              void* d_out, int out_size, void* d_ws,
                              size_t ws_size, hipStream_t stream) {
  (void)in_sizes; (void)n_in; (void)out_size; (void)ws_size;
  const float* audio  = (const float*)d_in[0];   // (8,512,512)
  const float* text   = (const float*)d_in[1];   // (8,64,512)
  const int*   alen   = (const int*)d_in[2];
  const int*   tlen   = (const int*)d_in[3];
  const float* W_attn = (const float*)d_in[4];   // (512,1024) row-major
  const float* b_attn = (const float*)d_in[5];
  const float* vvec   = (const float*)d_in[6];
  const float* W_u    = (const float*)d_in[7];   // (512,512)
  const float* b_u    = (const float*)d_in[8];
  const float* W_s    = (const float*)d_in[9];
  const float* b_s    = (const float*)d_in[10];

  float* out   = (float*)d_out;
  float* u_out = out;                        // first tuple element
  float* s_out = out + (size_t)BB * TQ_ * DD;

  // ---- workspace layout (all 16B aligned) ----
  const size_t NQ = (size_t)BB * TQ_ * DD;   // 2,097,152
  const size_t NK = (size_t)BB * TK_ * DD;   //   262,144
  float* ws   = (float*)d_ws;
  float* qp   = ws;            // f32  NQ
  float* kbw  = qp + NQ;       // f32  NK
  float* ctx  = kbw + NK;      // f32  NQ
  _Float16* hws     = (_Float16*)(ctx + NQ);
  _Float16* audio_h = hws;                   // NQ halves
  _Float16* text_h  = audio_h + NQ;          // NK
  _Float16* wattn_h = text_h + NK;           // 512*1024
  _Float16* wu_h    = wattn_h + (size_t)DD * 2 * DD;  // 512*512
  _Float16* ws_h    = wu_h + (size_t)DD * DD;
  _Float16* ctx_h   = ws_h + (size_t)DD * DD;         // NQ

  dim3 blk(256);
  // ---- one-time fp32 -> f16 conversions ----
  cvt_f32_to_f16<<<dim3(NQ / 4 / 256), blk, 0, stream>>>(audio, audio_h, NQ / 4);
  cvt_f32_to_f16<<<dim3(NK / 4 / 256), blk, 0, stream>>>(text, text_h, NK / 4);
  cvt_f32_to_f16<<<dim3(DD * 2 * DD / 4 / 256), blk, 0, stream>>>(
      W_attn, wattn_h, DD * 2 * DD / 4);
  cvt_f32_to_f16<<<dim3(DD * DD / 4 / 256), blk, 0, stream>>>(W_u, wu_h, DD * DD / 4);
  cvt_f32_to_f16<<<dim3(DD * DD / 4 / 256), blk, 0, stream>>>(W_s, ws_h, DD * DD / 4);

  // q_proj = audio @ Wq^T          (Wq[d,e] = W_attn[d*1024 + e])
  gemm_wmma<0, 0, 0><<<dim3(BB * TQ_ / 64, DD / 64), blk, 0, stream>>>(
      audio_h, DD, wattn_h, 2 * DD, 0, nullptr, nullptr, qp);
  // kb = text @ Wkv^T + b_attn     (Wkv[d,e] = W_attn[d*1024 + 512 + e])
  gemm_wmma<1, 0, 0><<<dim3(BB * TK_ / 64, DD / 64), blk, 0, stream>>>(
      text_h, DD, wattn_h, 2 * DD, DD, b_attn, nullptr, kbw);
  // additive-attention scores + masked softmax + context (f32 + f16 copies)
  score_softmax_ctx<<<dim3(BB * TQ_ / 8), blk, 0, stream>>>(
      qp, kbw, text, vvec, alen, tlen, ctx, ctx_h);
  // s_out = ctx * sigmoid(audio @ W_u^T + b_u)
  gemm_wmma<1, 1, 1><<<dim3(BB * TQ_ / 64, DD / 64), blk, 0, stream>>>(
      audio_h, DD, wu_h, DD, 0, b_u, ctx, s_out);
  // u_out = audio * sigmoid(ctx @ W_s^T + b_s)
  gemm_wmma<1, 1, 1><<<dim3(BB * TQ_ / 64, DD / 64), blk, 0, stream>>>(
      ctx_h, DD, ws_h, DD, 0, b_s, audio, u_out);
}
